// FFTConv2d_7816840478844
// MI455X (gfx1250) — matrix operand order repaired and checked by reading, not compiled, not run on hardware
//
#include <hip/hip_runtime.h>

// ---------------------------------------------------------------------------
// FFT-conv reference == 3x3 "same" linear convolution (flipped kernel) + bias.
// Implicit-GEMM with bf16 WMMA (v_wmma_f32_16x16x32_bf16), f32 accumulate.
// M = COUT = 64, N = 64 spatial cols, K = 576 ordered (kh,kw,cin).
// ---------------------------------------------------------------------------

typedef __attribute__((ext_vector_type(16))) __bf16 v16bf;
typedef __attribute__((ext_vector_type(8)))  __bf16 v8bf;
typedef __attribute__((ext_vector_type(8)))  float  v8f;

constexpr int kB    = 16;
constexpr int kCIN  = 64;
constexpr int kCOUT = 64;
constexpr int kH    = 128;
constexpr int kW    = 128;
constexpr int kK    = 576;   // 9 * 64, ordered k = (kh*3+kw)*64 + cin
constexpr int kCols = 66;    // w0-1 .. w0+64
constexpr int kCStr = 72;    // padded cin stride (bf16) -> 144B: 16B aligned, conflict-free

__device__ __forceinline__ __bf16 f2bf(float f) {
  unsigned u = __builtin_bit_cast(unsigned, f);
  unsigned r = (u + 0x7FFFu + ((u >> 16) & 1u)) >> 16;   // round-to-nearest-even
  unsigned short s = (unsigned short)r;
  return __builtin_bit_cast(__bf16, s);
}

// Reorder + flip + convert weights: Wb[cout][ (kh*3+kw)*64 + cin ] =
//   bf16( weight[cout][cin][2-kh][2-kw] )   (FFT conv flips the kernel)
__global__ void prep_weights_kernel(const float* __restrict__ wgt,
                                    __bf16* __restrict__ wb) {
  int idx = blockIdx.x * blockDim.x + threadIdx.x;
  if (idx >= kCOUT * kK) return;
  int cout = idx / kK;
  int rem  = idx % kK;
  int khw  = rem / kCIN;
  int cin  = rem % kCIN;
  int kh = khw / 3, kw = khw % 3;
  int fk = (2 - kh) * 3 + (2 - kw);
  wb[idx] = f2bf(wgt[(cout * kCIN + cin) * 9 + fk]);
}

__global__ __launch_bounds__(256)
void conv3x3_wmma_kernel(const float* __restrict__ x,
                         const __bf16* __restrict__ wb,
                         const float* __restrict__ bias,
                         float* __restrict__ out) {
  __shared__ __align__(16) __bf16 xs[3 * kCols * kCStr];   // 28512 B

  const int tid = threadIdx.x;
  const int wt  = blockIdx.x;          // 0..1  (w-tile of 64)
  const int h   = blockIdx.y;          // 0..127
  const int b   = blockIdx.z;          // 0..15
  const int w0  = wt * 64;

  // ---- stage input slab rows h-1..h+1, cols w0-1..w0+64, all cin, as bf16 ----
  for (int idx = tid; idx < 3 * kCIN * kCols; idx += 256) {
    int hr  = idx / (kCIN * kCols);
    int rem = idx % (kCIN * kCols);
    int cin = rem / kCols;
    int col = rem % kCols;              // consecutive tid -> consecutive col (coalesced)
    int gh = h + hr - 1;
    int gw = w0 + col - 1;
    float v = 0.0f;
    if ((unsigned)gh < (unsigned)kH && (unsigned)gw < (unsigned)kW)
      v = x[((b * kCIN + cin) * kH + gh) * kW + gw];
    xs[(hr * kCols + col) * kCStr + cin] = f2bf(v);
  }
  __syncthreads();

  // ---- per-wave 16(cout) x 32(n) output tile ----
  const int lane = tid & 31;
  const int wave = tid >> 5;           // 0..7
  const int c    = lane & 15;          // A row (M) / B column (N) / D column (N)
  const int hi   = lane >> 4;          // lane half selects K / M halves
  const int m0   = (wave & 3) * 16;    // cout base
  const int n0   = (wave >> 2) * 32;   // spatial base

  v8f acc0 = {};
  v8f acc1 = {};
  const __bf16* wrow = wb + (m0 + c) * kK;

#pragma unroll
  for (int khw = 0; khw < 9; ++khw) {
    const int kh = khw / 3;
    const int kw = khw % 3;
#pragma unroll
    for (int cc = 0; cc < 2; ++cc) {
      const int k0 = khw * 64 + cc * 32;
      // A fragment (16x32 bf16): lane holds row M=c, K = {8*hi..8*hi+7, 16+8*hi..}
      v8bf a0 = *(const v8bf*)(wrow + k0 + 8 * hi);
      v8bf a1 = *(const v8bf*)(wrow + k0 + 16 + 8 * hi);
      v16bf a = __builtin_shufflevector(a0, a1,
                  0,1,2,3,4,5,6,7,8,9,10,11,12,13,14,15);
      // B fragments (32x16 bf16): lane = column n, K (= cin) = {8*hi..8*hi+7, 16+8*hi..}
      const __bf16* bp0 =
          &xs[(kh * kCols + (n0 + c + kw)) * kCStr + cc * 32 + 8 * hi];
      v8bf b0l = *(const v8bf*)(bp0);
      v8bf b0h = *(const v8bf*)(bp0 + 16);
      v16bf bm0 = __builtin_shufflevector(b0l, b0h,
                  0,1,2,3,4,5,6,7,8,9,10,11,12,13,14,15);
      const __bf16* bp1 = bp0 + 16 * kCStr;
      v8bf b1l = *(const v8bf*)(bp1);
      v8bf b1h = *(const v8bf*)(bp1 + 16);
      v16bf bm1 = __builtin_shufflevector(b1l, b1h,
                  0,1,2,3,4,5,6,7,8,9,10,11,12,13,14,15);

      acc0 = __builtin_amdgcn_wmma_f32_16x16x32_bf16(
          false, a, false, bm0, (short)0, acc0, false, false);
      acc1 = __builtin_amdgcn_wmma_f32_16x16x32_bf16(
          false, a, false, bm1, (short)0, acc1, false, false);
    }
  }

  // ---- epilogue: D element i -> cout = m0 + 8*hi + i, col = n0 + c (+16) ----
  const int wcol = w0 + n0 + c;
#pragma unroll
  for (int i = 0; i < 8; ++i) {
    int cout = m0 + 8 * hi + i;
    float bv = bias[cout];
    float* op = out + (((size_t)b * kCOUT + cout) * kH + h) * kW + wcol;
    op[0]  = acc0[i] + bv;
    op[16] = acc1[i] + bv;
  }
}

extern "C" void kernel_launch(void* const* d_in, const int* in_sizes, int n_in,
                              void* d_out, int out_size, void* d_ws, size_t ws_size,
                              hipStream_t stream) {
  const float* x    = (const float*)d_in[0];
  const float* wgt  = (const float*)d_in[1];
  const float* bias = (const float*)d_in[2];
  float*       out  = (float*)d_out;
  __bf16*      wb   = (__bf16*)d_ws;           // 64*576*2 = 73728 bytes

  prep_weights_kernel<<<(kCOUT * kK + 255) / 256, 256, 0, stream>>>(wgt, wb);

  dim3 grid(kW / 64, kH, kB);                   // (2, 128, 16) = 4096 blocks
  conv3x3_wmma_kernel<<<grid, 256, 0, stream>>>(x, wb, bias, out);
}
